// res_block_4449586118760
// MI455X (gfx1250) — compile-verified, matrix operand order checked by reading, not run on hardware
//
#include <hip/hip_runtime.h>

// One-ring graph-conv residual block on gfx1250 (MI455X).
// Strategy: bf16 V_WMMA_F32_16X16X32_BF16 for both 896x128 GEMMs (f32 accum),
// gathers resolved as direct global_load_b128 of bf16 runs (one-ring data and
// the 0.45MB of weights live in the 192MB L2). BN stats use a deterministic
// two-pass reduction (no float atomics -> bitwise-stable across graph replays).
// Biases b1/b2 cancel exactly under BatchNorm mean subtraction and are skipped.

constexpr int NV = 163842;   // vertices
constexpr int CH = 128;      // channels
constexpr int KK = 896;      // 7 * CH
constexpr int STATS_ROWS = 512;
constexpr int NBLK_STATS = (NV + STATS_ROWS - 1) / STATS_ROWS;  // 321

typedef __attribute__((ext_vector_type(16))) __bf16 v16bf;
typedef __attribute__((ext_vector_type(8)))  float  v8f;

union ABFrag { uint4 u[2]; v16bf v; };   // 32 bytes: 16 bf16 per lane

__device__ __forceinline__ unsigned short f2bf(float f) {
  unsigned u = __builtin_bit_cast(unsigned, f);
  u = (u + 0x7FFFu + ((u >> 16) & 1u)) >> 16;   // round-to-nearest-even
  return (unsigned short)u;
}

__device__ __forceinline__ float lrelu(float v) { return v >= 0.f ? v : 0.2f * v; }

// ---------------------------------------------------------------- prep: f32->bf16
__global__ __launch_bounds__(256) void prep_kernel(
    const float* __restrict__ x, const float* __restrict__ W1,
    const float* __restrict__ W2, unsigned short* __restrict__ xb,
    unsigned short* __restrict__ w1b, unsigned short* __restrict__ w2b) {
  const long NX = (long)NV * CH;
  const long NW = (long)CH * KK;
  const long total = NX + 2 * NW;
  for (long i = blockIdx.x * (long)blockDim.x + threadIdx.x; i < total;
       i += (long)gridDim.x * blockDim.x) {
    if (i < NX)            xb[i]            = f2bf(x[i]);
    else if (i < NX + NW)  w1b[i - NX]      = f2bf(W1[i - NX]);
    else                   w2b[i - NX - NW] = f2bf(W2[i - NX - NW]);
  }
}

// ------------------------------------------------- gathered GEMM: h = gather(src) @ Wb^T
// Block: 256 threads (8 waves), covers 128 vertices x 128 channels.
// Wave: 32 vertices (2 M-tiles) x 64 channels (4 N-tiles) -> 8 f32 accumulators.
// A lane layout (16-bit 16x32, ISA 7.12.2): lane half h, element e maps to
// K = (e>>3)*16 + h*8 + (e&7)  -> two contiguous 8xbf16 (16B) runs per lane.
__global__ __launch_bounds__(256, 1) void gemm_kernel(
    const unsigned short* __restrict__ src,   // bf16 [NV][CH]
    const int* __restrict__ neigh,            // [NV*7]
    const unsigned short* __restrict__ Wb,    // bf16 [CH][KK] row-major
    float* __restrict__ hout) {               // f32 [NV][CH]
  const int lane = threadIdx.x & 31;
  const int wave = threadIdx.x >> 5;
  const int half = lane >> 4;
  const int lrow = lane & 15;
  const int mg = wave >> 1;                   // 0..3 : which 32-vertex group
  const int ng = wave & 1;                    // 0..1 : which 64-channel group
  const int blockBase = blockIdx.x * 128;

  const uint4* __restrict__ srcv = (const uint4*)src;  // 16 uint4 per vertex row
  const uint4* __restrict__ wv   = (const uint4*)Wb;   // 112 uint4 per co row

  int co[4];
#pragma unroll
  for (int nt = 0; nt < 4; ++nt) co[nt] = ng * 64 + nt * 16 + lrow;

  const int v0c = min(blockBase + mg * 32 + lrow,      NV - 1);
  const int v1c = min(blockBase + mg * 32 + 16 + lrow, NV - 1);

  const v8f vzero = {0.f, 0.f, 0.f, 0.f, 0.f, 0.f, 0.f, 0.f};
  v8f acc[2][4];
#pragma unroll
  for (int mi = 0; mi < 2; ++mi)
#pragma unroll
    for (int nt = 0; nt < 4; ++nt) acc[mi][nt] = vzero;

  for (int j = 0; j < 7; ++j) {               // one-ring neighbor slot
    const long a0row = (long)neigh[(long)v0c * 7 + j] * 16;
    const long a1row = (long)neigh[(long)v1c * 7 + j] * 16;
#pragma unroll
    for (int cc = 0; cc < 4; ++cc) {          // 32-wide K chunk within this neighbor
      const int k0 = j * 128 + cc * 32;
      const int q  = cc * 4 + half;           // (cc*32 + half*8)/8 in uint4 units
      ABFrag a0, a1;
      a0.u[0] = srcv[a0row + q];
      a0.u[1] = srcv[a0row + q + 2];
      a1.u[0] = srcv[a1row + q];
      a1.u[1] = srcv[a1row + q + 2];
#pragma unroll
      for (int nt = 0; nt < 4; ++nt) {
        ABFrag b;
        const long wb = (long)co[nt] * 112 + (k0 >> 3) + half;
        b.u[0] = wv[wb];
        b.u[1] = wv[wb + 2];
        acc[0][nt] = __builtin_amdgcn_wmma_f32_16x16x32_bf16(
            false, a0.v, false, b.v, (short)0, acc[0][nt], false, false);
        acc[1][nt] = __builtin_amdgcn_wmma_f32_16x16x32_bf16(
            false, a1.v, false, b.v, (short)0, acc[1][nt], false, false);
      }
    }
  }

  // C/D layout (ISA 7.12.2): lane half h, VGPR r -> (M = h*8 + r, N = lane&15)
#pragma unroll
  for (int mi = 0; mi < 2; ++mi) {
    const int vbase = blockBase + mg * 32 + mi * 16 + half * 8;
#pragma unroll
    for (int r = 0; r < 8; ++r) {
      const int v = vbase + r;
      if (v < NV) {
        float* __restrict__ row = hout + (long)v * CH;
#pragma unroll
        for (int nt = 0; nt < 4; ++nt) row[co[nt]] = acc[mi][nt][r];
      }
    }
  }
}

// ------------------------------------------- BN stats pass 1: per-block partials
__global__ __launch_bounds__(256) void stats_partial_kernel(
    const float* __restrict__ h, float* __restrict__ partials) {
  __shared__ float s1[256], s2[256];
  const int c = threadIdx.x & 127;
  const int g = threadIdx.x >> 7;   // 0/1: row parity within the block slab
  const int rbeg = blockIdx.x * STATS_ROWS + g;
  const int rend = min(blockIdx.x * STATS_ROWS + STATS_ROWS, NV);
  float s = 0.f, sq = 0.f;
  for (int r = rbeg; r < rend; r += 2) {
    float v = h[(long)r * CH + c];
    s += v; sq += v * v;
  }
  s1[threadIdx.x] = s; s2[threadIdx.x] = sq;
  __syncthreads();
  if (g == 0) {
    partials[(blockIdx.x * 2 + 0) * CH + c] = s1[c] + s1[c + 128];
    partials[(blockIdx.x * 2 + 1) * CH + c] = s2[c] + s2[c + 128];
  }
}

// ------------------------------------------- BN stats pass 2: mean / rstd (1 block)
__global__ void stats_final_kernel(const float* __restrict__ partials,
                                   float* __restrict__ meanv,
                                   float* __restrict__ rstdv) {
  const int c = threadIdx.x;        // 128 threads
  float s = 0.f, sq = 0.f;
  for (int b = 0; b < NBLK_STATS; ++b) {
    s  += partials[(b * 2 + 0) * CH + c];
    sq += partials[(b * 2 + 1) * CH + c];
  }
  const float invN = 1.f / (float)NV;
  const float mean = s * invN;
  const float var  = fmaxf(sq * invN - mean * mean, 0.f);
  meanv[c] = mean;
  rstdv[c] = rsqrtf(var + 1e-5f);
}

// --------------------------------- BN + LeakyReLU + bf16 convert (feeds layer 2)
__global__ __launch_bounds__(256) void bnapply_kernel(
    const float* __restrict__ h, const float* __restrict__ meanv,
    const float* __restrict__ rstdv, const float* __restrict__ gamma,
    const float* __restrict__ beta, unsigned short* __restrict__ outb) {
  const long total = (long)NV * CH;
  for (long i = blockIdx.x * (long)blockDim.x + threadIdx.x; i < total;
       i += (long)gridDim.x * blockDim.x) {
    const int c = (int)(i & (CH - 1));
    const float v = (h[i] - meanv[c]) * rstdv[c] * gamma[c] + beta[c];
    outb[i] = f2bf(lrelu(v));
  }
}

// --------------------------------- BN2 + residual + LeakyReLU -> f32 output
__global__ __launch_bounds__(256) void final_kernel(
    const float* __restrict__ h, const float* __restrict__ x,
    const float* __restrict__ meanv, const float* __restrict__ rstdv,
    const float* __restrict__ gamma, const float* __restrict__ beta,
    float* __restrict__ out) {
  const long total = (long)NV * CH;
  for (long i = blockIdx.x * (long)blockDim.x + threadIdx.x; i < total;
       i += (long)gridDim.x * blockDim.x) {
    const int c = (int)(i & (CH - 1));
    const float v = (h[i] - meanv[c]) * rstdv[c] * gamma[c] + beta[c] + x[i];
    out[i] = lrelu(v);
  }
}

extern "C" void kernel_launch(void* const* d_in, const int* in_sizes, int n_in,
                              void* d_out, int out_size, void* d_ws, size_t ws_size,
                              hipStream_t stream) {
  const float* x      = (const float*)d_in[0];
  const int*   neigh  = (const int*)d_in[1];
  const float* W1     = (const float*)d_in[2];
  // d_in[3] = b1, d_in[7] = b2: exactly cancelled by BatchNorm mean subtraction.
  const float* gamma1 = (const float*)d_in[4];
  const float* beta1  = (const float*)d_in[5];
  const float* W2     = (const float*)d_in[6];
  const float* gamma2 = (const float*)d_in[8];
  const float* beta2  = (const float*)d_in[9];
  float* out = (float*)d_out;

  // Workspace layout (256B aligned, ~168.6 MB total)
  char* ws = (char*)d_ws;
  unsigned short* xb  = (unsigned short*)(ws);                  // bf16 x      41,943,552 B
  unsigned short* h1n = (unsigned short*)(ws +  41943552ULL);   // bf16 act1   41,943,552 B
  float*          h   = (float*)         (ws +  83887104ULL);   // f32 h1/h2   83,887,104 B
  unsigned short* w1b = (unsigned short*)(ws + 167774208ULL);   // bf16 W1        229,376 B
  unsigned short* w2b = (unsigned short*)(ws + 168003584ULL);   // bf16 W2        229,376 B
  float*          prt = (float*)         (ws + 168232960ULL);   // stats partials 328,704 B
  float*          st  = (float*)         (ws + 168561664ULL);   // mean/rstd x2     2,048 B
  float* mean1 = st;        float* rstd1 = st + 128;
  float* mean2 = st + 256;  float* rstd2 = st + 384;

  const int gblocks = (NV + 127) / 128;   // 1281

  prep_kernel<<<4096, 256, 0, stream>>>(x, W1, W2, xb, w1b, w2b);

  gemm_kernel<<<gblocks, 256, 0, stream>>>(xb, neigh, w1b, h);
  stats_partial_kernel<<<NBLK_STATS, 256, 0, stream>>>(h, prt);
  stats_final_kernel<<<1, 128, 0, stream>>>(prt, mean1, rstd1);
  bnapply_kernel<<<4096, 256, 0, stream>>>(h, mean1, rstd1, gamma1, beta1, h1n);

  gemm_kernel<<<gblocks, 256, 0, stream>>>(h1n, neigh, w2b, h);
  stats_partial_kernel<<<NBLK_STATS, 256, 0, stream>>>(h, prt);
  stats_final_kernel<<<1, 128, 0, stream>>>(prt, mean2, rstd2);
  final_kernel<<<4096, 256, 0, stream>>>(h, x, mean2, rstd2, gamma2, beta2, out);
}
